// GATEncoder_67980742361713
// MI455X (gfx1250) — matrix operand order repaired and checked by reading, not compiled, not run on hardware
//
#include <hip/hip_runtime.h>
#include <hip/hip_bf16.h>

typedef __attribute__((ext_vector_type(16))) __bf16 v16bf;
typedef __attribute__((ext_vector_type(8)))  __bf16 v8bf;
typedef __attribute__((ext_vector_type(8)))  float  v8f;

#define LRELU_SLOPE 0.2f

// ---------------- elementwise helpers ----------------
__global__ void k_f32_to_bf16(const float* __restrict__ in, __bf16* __restrict__ out, int n) {
  int i = blockIdx.x * blockDim.x + threadIdx.x;
  if (i < n) out[i] = (__bf16)in[i];
}

__global__ void k_fill_f32(float* __restrict__ p, float v, int n) {
  int i = blockIdx.x * blockDim.x + threadIdx.x;
  if (i < n) p[i] = v;
}

// ---------------- pack weight matrix into WMMA B-fragment order ----------------
// B[K,N] f32 row-major -> bf16 fragments. Fragment f = kt*ntn + tn holds, per lane,
// 16 bf16: b[j] = B[kt*32 + half*16 + j][tn*16 + l16]  (contiguous k order, same as A in the GEMM).
// Stored contiguously: Bp[(f*32 + lane)*16 .. +15]  => GEMM reads 32B/lane (2x b128).
__global__ void k_pack_b(const float* __restrict__ B, __bf16* __restrict__ Bp, int N, int K) {
  const int lane = threadIdx.x & 31;
  const int f    = blockIdx.x * (blockDim.x >> 5) + (threadIdx.x >> 5);
  const int ntn  = N >> 4;
  const int nk   = K >> 5;
  if (f >= nk * ntn) return;
  const int kt = f / ntn, tn = f % ntn;
  const int half = lane >> 4, l16 = lane & 15;
  const float* bcol = B + (kt * 32 + half * 16) * N + tn * 16 + l16;
  v16bf b;
  #pragma unroll
  for (int j = 0; j < 16; ++j) b[j] = (__bf16)bcol[j * N];
  *(v16bf*)(Bp + ((size_t)f * 32 + lane) * 16) = b;
}

// ---------------- WMMA GEMM: C[M,N] = A[M,K](bf16) * Bp(packed), f32 out ----------------
// One wave per 16x16 tile. A: lane l16 -> row M; elems 0..15 = K[half*16..half*16+15]
// (matches packed B). C: VGPR i -> row half*8+i, col l16.
__global__ void k_wmma_gemm(const __bf16* __restrict__ A, const __bf16* __restrict__ Bp,
                            float* __restrict__ C, int M, int N, int K) {
  const int ntn  = N >> 4;
  const int ntm  = M >> 4;
  const int nk   = K >> 5;
  const int tile = blockIdx.x * (blockDim.x >> 5) + (threadIdx.x >> 5);
  if (tile >= ntm * ntn) return;              // wave-uniform exit (EXEC stays all-ones)
  const int tm   = tile / ntn;
  const int tn   = tile % ntn;
  const int lane = threadIdx.x & 31;
  const int half = lane >> 4;
  const int l16  = lane & 15;

  v8f acc = {};
  const __bf16* arow = A + (tm * 16 + l16) * K;          // 32-bit offset math
  const __bf16* bbase = Bp + ((size_t)tn * 32 + lane) * 16;
  for (int kt = 0; kt < nk; ++kt) {
    v16bf a = *(const v16bf*)(arow + kt * 32 + half * 16);     // 32B-aligned
    v16bf b = *(const v16bf*)(bbase + (size_t)kt * ntn * 32 * 16);
    acc = __builtin_amdgcn_wmma_f32_16x16x32_bf16(false, a, false, b, (short)0, acc,
                                                  false, false);
  }
  const int col = tn * 16 + l16;
  float* crow = C + (tm * 16 + half * 8) * N + col;
  #pragma unroll
  for (int i = 0; i < 8; ++i) crow[i * N] = acc[i];
}

// ---------------- attention coefficients: a_s[n,h] = <H[n,h,:], att_src[h,:]> ----------------
__global__ void k_attn_coef(const float* __restrict__ H, const float* __restrict__ att_src,
                            const float* __restrict__ att_dst, float* __restrict__ a_s,
                            float* __restrict__ a_d, int N, int Hh, int O) {
  int idx = blockIdx.x * blockDim.x + threadIdx.x;
  if (idx >= N * Hh) return;
  int n = idx / Hh, h = idx - n * Hh;
  const float* hp = H + (size_t)n * Hh * O + h * O;
  const float* as = att_src + h * O;
  const float* ad = att_dst + h * O;
  float s = 0.f, d = 0.f;
  for (int o = 0; o < O; ++o) { s += hp[o] * as[o]; d += hp[o] * ad[o]; }
  a_s[idx] = s; a_d[idx] = d;
}

// ---------------- edge helpers ----------------
__device__ __forceinline__ float lrelu(float v) { return v > 0.f ? v : LRELU_SLOPE * v; }

__device__ __forceinline__ void atomicMaxF(float* addr, float v) {
  // signed max for v>=0, unsigned min for v<0 (valid with -inf init)
  if (v >= 0.f) atomicMax((int*)addr, __float_as_int(v));
  else          atomicMin((unsigned int*)addr, __float_as_uint(v));
}

__global__ void k_edge_max(const int* __restrict__ es, const int* __restrict__ ed,
                           int E0, int Et, int Hh,
                           const float* __restrict__ a_s, const float* __restrict__ a_d,
                           float* __restrict__ emax) {
  int idx = blockIdx.x * blockDim.x + threadIdx.x;
  if (idx >= Et * Hh) return;
  int e = idx / Hh, h = idx - e * Hh;
  int s = (e < E0) ? es[e] : (e - E0);     // appended self-loops
  int d = (e < E0) ? ed[e] : (e - E0);
  float v = lrelu(a_s[s * Hh + h] + a_d[d * Hh + h]);
  atomicMaxF(&emax[d * Hh + h], v);
}

__global__ void k_edge_denom(const int* __restrict__ es, const int* __restrict__ ed,
                             int E0, int Et, int Hh,
                             const float* __restrict__ a_s, const float* __restrict__ a_d,
                             const float* __restrict__ emax, float* __restrict__ denom) {
  int idx = blockIdx.x * blockDim.x + threadIdx.x;
  if (idx >= Et * Hh) return;
  int e = idx / Hh, h = idx - e * Hh;
  int s = (e < E0) ? es[e] : (e - E0);
  int d = (e < E0) ? ed[e] : (e - E0);
  int dh = d * Hh + h;
  float v = lrelu(a_s[s * Hh + h] + a_d[dh]);
  atomicAdd(&denom[dh], __expf(v - emax[dh]));
}

// One wave per (edge, head); lane = output channel o (O fixed at 32).
// Atomic adds across a wave hit 32 consecutive floats -> coalesced global_atomic_add_f32.
__global__ void k_edge_agg(const int* __restrict__ es, const int* __restrict__ ed,
                           int E0, long long total, int Hh,
                           const float* __restrict__ a_s, const float* __restrict__ a_d,
                           const float* __restrict__ emax, const float* __restrict__ denom,
                           const float* __restrict__ H, float* __restrict__ agg) {
  long long idx = (long long)blockIdx.x * blockDim.x + threadIdx.x;
  if (idx >= total) return;                 // total = Et*Hh*32
  int o = (int)(idx & 31);
  long long eh = idx >> 5;                  // wave-uniform
  int h = (int)(eh % Hh);
  int e = (int)(eh / Hh);
  int s = (e < E0) ? es[e] : (e - E0);
  int d = (e < E0) ? ed[e] : (e - E0);
  int dh = d * Hh + h;
  float v = lrelu(a_s[s * Hh + h] + a_d[dh]);
  float alpha = __expf(v - emax[dh]) / denom[dh];
  atomicAdd(&agg[(size_t)d * Hh * 32 + h * 32 + o],
            H[(size_t)s * Hh * 32 + h * 32 + o] * alpha);
}

// ---------------- finalizers ----------------
__global__ void k_finalize1(const float* __restrict__ agg, const float* __restrict__ b1,
                            __bf16* __restrict__ out_bf, int n, int C) {
  int i = blockIdx.x * blockDim.x + threadIdx.x;
  if (i >= n) return;
  float v = agg[i] + b1[i % C];
  v = (v > 0.f) ? v : (__expf(v) - 1.f);    // ELU
  out_bf[i] = (__bf16)v;
}

__global__ void k_finalize2(const float* __restrict__ agg, const float* __restrict__ b2,
                            float* __restrict__ out, int n, int C) {
  int i = blockIdx.x * blockDim.x + threadIdx.x;
  if (i >= n) return;
  out[i] = agg[i] + b2[i % C];
}

// ---------------- launch ----------------
extern "C" void kernel_launch(void* const* d_in, const int* in_sizes, int n_in,
                              void* d_out, int out_size, void* d_ws, size_t ws_size,
                              hipStream_t stream) {
  const int IN_DIM = 64, HID = 32, HEADS = 4, OUT = 32;
  const int C1 = HEADS * HID;                 // 128
  const int N  = in_sizes[0] / IN_DIM;        // 50000
  const int E0 = in_sizes[1] / 2;             // 800000
  const int Et = E0 + N;                      // + self loops

  const float* x    = (const float*)d_in[0];
  const int*   esrc = (const int*)d_in[1];
  const int*   edst = esrc + E0;
  const float* W1   = (const float*)d_in[2];
  const float* as1  = (const float*)d_in[3];
  const float* ad1  = (const float*)d_in[4];
  const float* b1   = (const float*)d_in[5];
  const float* W2   = (const float*)d_in[6];
  const float* as2  = (const float*)d_in[7];
  const float* ad2  = (const float*)d_in[8];
  const float* b2   = (const float*)d_in[9];

  // ---- workspace bump allocator (layer-2 scratch overlays layer-1 scratch) ----
  uint8_t* ws = (uint8_t*)d_ws;
  size_t off = 0;
  auto alloc = [&](size_t bytes) -> void* {
    void* p = ws + off;
    off = (off + bytes + 255) & ~(size_t)255;
    return p;
  };
  const int frags1 = (IN_DIM / 32) * (C1 / 16);   // 2*8  = 16
  const int frags2 = (C1 / 32) * (OUT / 16);      // 4*2  = 8
  // persistent
  __bf16* Xbf  = (__bf16*)alloc((size_t)N * IN_DIM * 2);
  __bf16* Bp1  = (__bf16*)alloc((size_t)frags1 * 32 * 16 * 2);
  __bf16* Bp2  = (__bf16*)alloc((size_t)frags2 * 32 * 16 * 2);
  __bf16* H1bf = (__bf16*)alloc((size_t)N * C1 * 2);    // elu(layer1 out), bf16
  const size_t scratch0 = off;
  // layer-1 scratch
  float* H1    = (float*)alloc((size_t)N * C1 * 4);
  float* aS1   = (float*)alloc((size_t)N * HEADS * 4);
  float* aD1   = (float*)alloc((size_t)N * HEADS * 4);
  float* emax1 = (float*)alloc((size_t)N * HEADS * 4);
  float* den1  = (float*)alloc((size_t)N * HEADS * 4);
  float* agg1  = (float*)alloc((size_t)N * C1 * 4);
  // layer-2 scratch overlays layer-1 scratch
  off = scratch0;
  float* H2    = (float*)alloc((size_t)N * OUT * 4);
  float* aS2   = (float*)alloc((size_t)N * 4);
  float* aD2   = (float*)alloc((size_t)N * 4);
  float* emax2 = (float*)alloc((size_t)N * 4);
  float* den2  = (float*)alloc((size_t)N * 4);
  float* agg2  = (float*)alloc((size_t)N * OUT * 4);
  (void)ws_size; (void)n_in;

  const int B = 256;
  auto g = [&](long long n) { return (int)((n + B - 1) / B); };

  // ---- convert X, pack weights into fragment order (fused f32->bf16) ----
  k_f32_to_bf16<<<g((long long)N * IN_DIM), B, 0, stream>>>(x, Xbf, N * IN_DIM);
  k_pack_b<<<(frags1 + 3) / 4, 128, 0, stream>>>(W1, Bp1, C1, IN_DIM);
  k_pack_b<<<(frags2 + 3) / 4, 128, 0, stream>>>(W2, Bp2, OUT, C1);

  // ---- layer 1 ----
  {
    int tiles = (N / 16) * (C1 / 16);
    k_wmma_gemm<<<(tiles + 3) / 4, 128, 0, stream>>>(Xbf, Bp1, H1, N, C1, IN_DIM);
  }
  k_attn_coef<<<g((long long)N * HEADS), B, 0, stream>>>(H1, as1, ad1, aS1, aD1, N, HEADS, HID);
  k_fill_f32<<<g((long long)N * HEADS), B, 0, stream>>>(emax1, -__builtin_inff(), N * HEADS);
  k_fill_f32<<<g((long long)N * HEADS), B, 0, stream>>>(den1, 0.f, N * HEADS);
  k_fill_f32<<<g((long long)N * C1),    B, 0, stream>>>(agg1, 0.f, N * C1);
  {
    long long w = (long long)Et * HEADS;
    k_edge_max  <<<g(w), B, 0, stream>>>(esrc, edst, E0, Et, HEADS, aS1, aD1, emax1);
    k_edge_denom<<<g(w), B, 0, stream>>>(esrc, edst, E0, Et, HEADS, aS1, aD1, emax1, den1);
    k_edge_agg  <<<g(w * 32), B, 0, stream>>>(esrc, edst, E0, w * 32, HEADS,
                                              aS1, aD1, emax1, den1, H1, agg1);
  }
  k_finalize1<<<g((long long)N * C1), B, 0, stream>>>(agg1, b1, H1bf, N * C1, C1);

  // ---- layer 2 ----
  {
    int tiles = (N / 16) * (OUT / 16);
    k_wmma_gemm<<<(tiles + 3) / 4, 128, 0, stream>>>(H1bf, Bp2, H2, N, OUT, C1);
  }
  k_attn_coef<<<g(N), B, 0, stream>>>(H2, as2, ad2, aS2, aD2, N, 1, OUT);
  k_fill_f32<<<g(N), B, 0, stream>>>(emax2, -__builtin_inff(), N);
  k_fill_f32<<<g(N), B, 0, stream>>>(den2, 0.f, N);
  k_fill_f32<<<g((long long)N * OUT), B, 0, stream>>>(agg2, 0.f, N * OUT);
  {
    long long w = (long long)Et;
    k_edge_max  <<<g(w), B, 0, stream>>>(esrc, edst, E0, Et, 1, aS2, aD2, emax2);
    k_edge_denom<<<g(w), B, 0, stream>>>(esrc, edst, E0, Et, 1, aS2, aD2, emax2, den2);
    k_edge_agg  <<<g(w * 32), B, 0, stream>>>(esrc, edst, E0, w * 32, 1,
                                              aS2, aD2, emax2, den2, H2, agg2);
  }
  k_finalize2<<<g((long long)N * OUT), B, 0, stream>>>(agg2, b2, (float*)d_out, N * OUT, OUT);
}
